// ConstraintCRF_86689619903492
// MI455X (gfx1250) — matrix operand order repaired and checked by reading, not compiled, not run on hardware
//
#include <hip/hip_runtime.h>
#include <hip/hip_bf16.h>

// CRF log-likelihood on gfx1250.
// Denominator recurrence recast as f16 WMMA GEMM in probability domain:
//   new_alpha = rowmax + log( exp(alpha-rowmax) @ exp(T) ) + logit_t
// exp(T) in [~0.7,1.4] (T ~ N(0,1/K)) -> safe in f16; f32 accumulation.

typedef __attribute__((ext_vector_type(16))) _Float16 v16h;
typedef __attribute__((ext_vector_type(8)))  _Float16 v8h;
typedef __attribute__((ext_vector_type(4)))  _Float16 v4h;
typedef __attribute__((ext_vector_type(8)))  float    v8f;
typedef __attribute__((ext_vector_type(4)))  float    v4f;

#define CRF_B 128
#define CRF_T 512
#define CRF_K 256

// ---------------------------------------------------------------------------
// Denominator: one workgroup per 16-batch tile, 16 waves (512 thr).
// Wave w owns output columns [16w, 16w+16). expT B-fragments stay in VGPRs
// across the whole T loop. alpha (f32) and P (f16) live in LDS.
// ---------------------------------------------------------------------------
__global__ void __launch_bounds__(512, 1)
crf_denominator(const float* __restrict__ logits,
                const unsigned char* __restrict__ mask,
                const float* __restrict__ trans,
                const float* __restrict__ startT,
                const float* __restrict__ endT,
                float* __restrict__ den)
{
    constexpr int ASTR = 260;   // f32 alpha row stride: 16B-aligned, bank-skewed
    constexpr int PSTR = 264;   // f16 P row stride: 16B-aligned rows
    __shared__ __align__(16) float         s_alpha[16 * ASTR];
    __shared__ __align__(16) _Float16      s_p[16 * PSTR];
    __shared__ __align__(16) float         s_rowmax[16];
    __shared__ __align__(16) unsigned char s_mk[16];

    const int tid  = threadIdx.x;
    const int wave = tid >> 5;
    const int lane = tid & 31;
    const int hi   = lane >> 4;               // half-wave select (0/1)
    const int b0   = blockIdx.x * 16;         // batch tile base
    const int n0   = wave * 16;               // this wave's column tile
    const int colg = n0 + (lane & 15);        // this lane's global column

    // --- B fragments: expT[k, colg], k = c*32 + 16*hi + e, contiguous (same order as A) ------
    // VGPR pair v holds rows K = 2v,2v+1 (lanes 0-15) / 16+2v,16+2v+1 (16-31).
    v16h Bf[8];
#pragma unroll
    for (int c = 0; c < 8; ++c) {
#pragma unroll
        for (int e = 0; e < 16; ++e) {
            int k = c * 32 + hi * 16 + e;
            Bf[c][e] = (_Float16)__expf(trans[k * CRF_K + colg]);
        }
    }

    // --- alpha0 = start_transitions + logits[:,0,:] -------------------------
    for (int i = tid; i < 16 * CRF_K; i += 512) {
        int r = i >> 8, k = i & 255;
        s_alpha[r * ASTR + k] =
            startT[k] + logits[(size_t)(b0 + r) * CRF_T * CRF_K + k];
    }

    for (int t = 1; t < CRF_T; ++t) {
        __syncthreads();                      // alpha writes from prev step done

        // stage this step's mask bytes for the batch tile (read after 3rd bar)
        if (tid < 16) s_mk[tid] = mask[(b0 + tid) * CRF_T + t];

        // row max (wave w reduces row w) — vectorized reads
        float m = -3.4e38f;
        {
            const v4f* arow = (const v4f*)&s_alpha[wave * ASTR];
            for (int q = lane; q < CRF_K / 4; q += 32) {
                v4f a = arow[q];
                m = fmaxf(m, fmaxf(fmaxf(a.x, a.y), fmaxf(a.z, a.w)));
            }
        }
#pragma unroll
        for (int off = 16; off; off >>= 1)
            m = fmaxf(m, __shfl_xor(m, off, 32));
        if (lane == 0) s_rowmax[wave] = m;
        __syncthreads();

        // P = exp(alpha - rowmax) -> f16 in LDS (b128 load / b64 store)
        for (int i = tid; i < 16 * (CRF_K / 4); i += 512) {
            int r = i >> 6, q = i & 63;
            v4f  a  = *(const v4f*)&s_alpha[r * ASTR + q * 4];
            float mr = s_rowmax[r];
            v4h  p;
            p.x = (_Float16)__expf(a.x - mr);
            p.y = (_Float16)__expf(a.y - mr);
            p.z = (_Float16)__expf(a.z - mr);
            p.w = (_Float16)__expf(a.w - mr);
            *(v4h*)&s_p[r * PSTR + q * 4] = p;
        }
        __syncthreads();

        // Load all 8 A fragments into distinct registers (bulk ds_load_b128),
        // then run the WMMA chain on two accumulators to shorten RAW chains.
        v16h afr[8];
#pragma unroll
        for (int c = 0; c < 8; ++c) {
            int row = lane & 15;
            int kb  = c * 32 + hi * 16;
            v8h lo = *(const v8h*)&s_p[row * PSTR + kb];
            v8h h8 = *(const v8h*)&s_p[row * PSTR + kb + 8];
            afr[c] = __builtin_shufflevector(lo, h8, 0, 1, 2, 3, 4, 5, 6, 7,
                                             8, 9, 10, 11, 12, 13, 14, 15);
        }
        v8f acc0 = {}, acc1 = {};
#pragma unroll
        for (int c = 0; c < 8; c += 2) {
            acc0 = __builtin_amdgcn_wmma_f32_16x16x32_f16(
                false, afr[c], false, Bf[c], (short)0, acc0, false, false);
            acc1 = __builtin_amdgcn_wmma_f32_16x16x32_f16(
                false, afr[c + 1], false, Bf[c + 1], (short)0, acc1, false, false);
        }
        v8f acc = acc0 + acc1;

        // prefetch logits 8 steps ahead (global_prefetch_b8)
        if (t + 8 < CRF_T)
            __builtin_prefetch(&logits[(size_t)(b0 + (lane & 15)) * CRF_T * CRF_K +
                                       (size_t)(t + 8) * CRF_K + n0], 0, 1);

        // Vector-load the 8 rowmaxes / 8 mask bytes this lane needs
        // (rows hi*8 .. hi*8+7): 2x ds_load_b128 + 2x ds_load_b32 total.
        v4f rm0 = *(const v4f*)&s_rowmax[hi * 8];
        v4f rm1 = *(const v4f*)&s_rowmax[hi * 8 + 4];
        unsigned int mk0 = *(const unsigned int*)&s_mk[hi * 8];
        unsigned int mk1 = *(const unsigned int*)&s_mk[hi * 8 + 4];

        // writeback: C layout -> lane element c is row c + 8*hi, col = colg
#pragma unroll
        for (int c = 0; c < 8; ++c) {
            int r = c + hi * 8;
            float rm = (c < 4) ? rm0[c] : rm1[c - 4];
            unsigned int mkb = (((c < 4) ? mk0 : mk1) >> ((c & 3) * 8)) & 0xFFu;
            float v = rm + __logf(acc[c]) +
                      logits[((size_t)(b0 + r) * CRF_T + t) * CRF_K + colg];
            if (mkb)
                s_alpha[r * ASTR + colg] = v;
        }
    }

    __syncthreads();
    // den[b] = logsumexp_j(alpha[b,j] + endT[j]); wave w handles row w
    float m = -3.4e38f;
    for (int k = lane; k < CRF_K; k += 32)
        m = fmaxf(m, s_alpha[wave * ASTR + k] + endT[k]);
#pragma unroll
    for (int off = 16; off; off >>= 1)
        m = fmaxf(m, __shfl_xor(m, off, 32));
    float s = 0.f;
    for (int k = lane; k < CRF_K; k += 32)
        s += __expf(s_alpha[wave * ASTR + k] + endT[k] - m);
#pragma unroll
    for (int off = 16; off; off >>= 1)
        s += __shfl_xor(s, off, 32);
    if (lane == 0) den[b0 + wave] = m + __logf(s);
}

// ---------------------------------------------------------------------------
// Numerator: one block per batch row; gather-sum over T.
// ---------------------------------------------------------------------------
__global__ void __launch_bounds__(256)
crf_numerator(const float* __restrict__ logits,
              const int* __restrict__ tags,
              const unsigned char* __restrict__ mask,
              const float* __restrict__ trans,
              const float* __restrict__ startT,
              const float* __restrict__ endT,
              float* __restrict__ num)
{
    const int b = blockIdx.x, tid = threadIdx.x;
    __shared__ float red[256];

    float s = 0.f;
    for (int t = 1 + tid; t < CRF_T; t += 256) {          // transition terms
        float mf = mask[b * CRF_T + t] ? 1.f : 0.f;
        s += trans[tags[b * CRF_T + t - 1] * CRF_K + tags[b * CRF_T + t]] * mf;
    }
    for (int t = tid; t < CRF_T - 1; t += 256) {          // emission terms
        float mf = mask[b * CRF_T + t] ? 1.f : 0.f;
        s += logits[(size_t)b * CRF_T * CRF_K + (size_t)t * CRF_K +
                    tags[b * CRF_T + t]] * mf;
    }
    red[tid] = s;
    __syncthreads();
    for (int o = 128; o > 0; o >>= 1) {
        if (tid < o) red[tid] += red[tid + o];
        __syncthreads();
    }
    if (tid == 0) {
        int cnt = 0;
        for (int t = 0; t < CRF_T; ++t) cnt += mask[b * CRF_T + t] ? 1 : 0;
        int last = cnt - 1;
        int lt   = tags[b * CRF_T + last];
        float sc = red[0] + startT[tags[b * CRF_T]] + endT[lt];
        float mfl = mask[b * CRF_T + CRF_T - 1] ? 1.f : 0.f;
        sc += logits[(size_t)b * CRF_T * CRF_K +
                     (size_t)(CRF_T - 1) * CRF_K + lt] * mfl;
        num[b] = sc;
    }
}

// ---------------------------------------------------------------------------
// Final scalar: sum_b (num[b] - den[b])
// ---------------------------------------------------------------------------
__global__ void __launch_bounds__(128)
crf_final(const float* __restrict__ den, const float* __restrict__ num,
          float* __restrict__ out)
{
    __shared__ float red[128];
    int t = threadIdx.x;
    red[t] = num[t] - den[t];
    __syncthreads();
    for (int o = 64; o > 0; o >>= 1) {
        if (t < o) red[t] += red[t + o];
        __syncthreads();
    }
    if (t == 0) out[0] = red[0];
}

extern "C" void kernel_launch(void* const* d_in, const int* in_sizes, int n_in,
                              void* d_out, int out_size, void* d_ws, size_t ws_size,
                              hipStream_t stream) {
    const float*         logits = (const float*)d_in[0];
    const int*           tags   = (const int*)d_in[1];
    const unsigned char* mask   = (const unsigned char*)d_in[2];   // jnp bool = 1B
    const float*         trans  = (const float*)d_in[3];
    const float*         startT = (const float*)d_in[4];
    const float*         endT   = (const float*)d_in[5];

    float* den = (float*)d_ws;          // 128 floats
    float* num = den + CRF_B;           // 128 floats

    crf_denominator<<<CRF_B / 16, 512, 0, stream>>>(logits, mask, trans,
                                                    startT, endT, den);
    crf_numerator<<<CRF_B, 256, 0, stream>>>(logits, tags, mask, trans,
                                             startT, endT, num);
    crf_final<<<1, 128, 0, stream>>>(den, num, (float*)d_out);
}
